// QuantumInspiredStateOptimizer_89154931130749
// MI455X (gfx1250) — matrix operand order repaired and checked by reading, not compiled, not run on hardware
//
#include <hip/hip_runtime.h>
#include <cmath>

// ---------------------------------------------------------------------------
// QuantumInspiredStateOptimizer for MI455X (gfx1250)
//
// Per step (100 sequential steps):
//   rowstats   : lin/mixer/pen/sum_old per row; zero quad          (~16MB read)
//   gemm_quad  : quad[b] = sum_n state[b,n]*(state@Q)[b,n]
//                bf16 WMMA 16x16x32, Q pre-converted+transposed to bf16 (32MB,
//                L2-resident), tiles staged via TDM tensor_load_to_lds
//   hamiltonian: H[b], evo[b]
//   candidate  : cand = evo*state + noise; sum_new, pen_cand per row
//   accept     : accept_prob, accept flag, energy_history[step]
//   commit     : state = accept ? cand : state; refresh bf16 state copy
// ---------------------------------------------------------------------------

#define NSTEPS 100
#define BATCH  512
#define DIM    4096

typedef __attribute__((ext_vector_type(16))) __bf16       v16bf;
typedef __attribute__((ext_vector_type(8)))  float        v8f;
typedef __attribute__((ext_vector_type(4)))  unsigned int v4u;
typedef __attribute__((ext_vector_type(8)))  int          v8i;
typedef __attribute__((ext_vector_type(4)))  int          v4i;

#if defined(__HIP_DEVICE_COMPILE__) && __has_builtin(__builtin_amdgcn_tensor_load_to_lds) && __has_builtin(__builtin_amdgcn_s_wait_tensorcnt)
#define HAS_TDM 1
#else
#define HAS_TDM 0
#endif

// ---------------- small device helpers ----------------

__device__ __forceinline__ unsigned short f2bf(float f) {
  unsigned u = __float_as_uint(f);
  unsigned r = 0x7FFFu + ((u >> 16) & 1u);
  return (unsigned short)((u + r) >> 16);
}

__device__ __forceinline__ unsigned hash3(unsigned a, unsigned b, unsigned c) {
  unsigned x = a * 0x9E3779B1u ^ b * 0x85EBCA77u ^ c * 0xC2B2AE3Du;
  x ^= x >> 16; x *= 0x7FEB352Du;
  x ^= x >> 15; x *= 0x846CA68Bu;
  x ^= x >> 16;
  return x;
}

__device__ __forceinline__ float gauss01(unsigned seed) {
  unsigned u1 = hash3(seed, 0x68BC21EBu, 0x02E5BE93u);
  unsigned u2 = hash3(seed, 0x967A889Bu, 0x51633E2Du);
  float f1 = ((float)u1 + 1.0f) * 2.3283064e-10f;  // (0,1]
  float f2 = (float)u2 * 2.3283064e-10f;           // [0,1)
  return sqrtf(-2.0f * logf(f1)) * cosf(6.28318530718f * f2);
}

template <int N>
__device__ __forceinline__ float block_reduce(float v, float* sm) {
  int tid = threadIdx.x;
  sm[tid] = v;
  __syncthreads();
#pragma unroll
  for (int s = N >> 1; s > 0; s >>= 1) {
    if (tid < s) sm[tid] += sm[tid + s];
    __syncthreads();
  }
  float r = sm[0];
  __syncthreads();
  return r;
}

struct Two128 { uint4 lo, hi; };
__device__ __forceinline__ v16bf mk_frag(uint4 lo, uint4 hi) {
  Two128 t{lo, hi};
  return __builtin_bit_cast(v16bf, t);
}

#if HAS_TDM
// Issue a 2-D TDM tile load: tile_k contiguous bf16 elems x tile_rows rows,
// row pitch = stride_elems, into LDS at lds_off (packed row-major).
// This toolchain exposes the 6-arg builtin:
//   (v4u group0, v8i group1, v4i group2, v4i group3, v8i group4, i32 cpol)
__device__ __forceinline__ void tdm_load_2d(unsigned lds_off, const void* gaddr,
                                            unsigned tile_k, unsigned tile_rows,
                                            unsigned tensor_d0, unsigned tensor_d1,
                                            unsigned stride_elems) {
  unsigned long long ga = (unsigned long long)(size_t)gaddr;
  v4u g0;
  g0[0] = 1u;                                                   // count=1, user D#
  g0[1] = lds_off;                                              // lds_addr (bytes)
  g0[2] = (unsigned)(ga & 0xFFFFFFFFu);                         // global_addr[31:0]
  g0[3] = (unsigned)((ga >> 32) & 0x01FFFFFFu) | (2u << 30);    // addr[56:32] | type=2
  v8i g1;
  g1[0] = (int)(1u << 16);                                      // data_size=1 (2 bytes)
  g1[1] = (int)((tensor_d0 & 0xFFFFu) << 16);                   // tensor_dim0[15:0]
  g1[2] = (int)((tensor_d0 >> 16) | ((tensor_d1 & 0xFFFFu) << 16));
  g1[3] = (int)((tensor_d1 >> 16) | (tile_k << 16));            // tile_dim0
  g1[4] = (int)tile_rows;                                       // tile_dim1 (tile_dim2=0)
  g1[5] = (int)stride_elems;                                    // tensor_dim0_stride[31:0]
  g1[6] = 0;
  g1[7] = 0;
  v4i g2 = {1, 0, 0, 0};                                        // tensor_dim2=1, rest 0
  v4i g3 = {0, 0, 0, 0};
  v8i g4 = {0, 0, 0, 0, 0, 0, 0, 0};                            // unused trailing group
  __builtin_amdgcn_tensor_load_to_lds(g0, g1, g2, g3, g4, 0);
}
#endif

// ---------------- kernels ----------------

// One-time: Qt[n][k] = bf16(Q[k][n])   (transpose + convert via LDS tile)
__global__ void qt_convert_kernel(const float* __restrict__ Q, unsigned short* __restrict__ Qt) {
  __shared__ float tile[32][33];
  int bx = blockIdx.x, by = blockIdx.y;      // bx: n-tile, by: k-tile
  int tx = threadIdx.x, ty = threadIdx.y;    // 32 x 8
#pragma unroll
  for (int i = 0; i < 4; ++i)
    tile[ty + 8 * i][tx] = Q[(size_t)(by * 32 + ty + 8 * i) * DIM + bx * 32 + tx];
  __syncthreads();
#pragma unroll
  for (int i = 0; i < 4; ++i)
    Qt[(size_t)(bx * 32 + ty + 8 * i) * DIM + by * 32 + tx] = f2bf(tile[tx][ty + 8 * i]);
}

__global__ void init_state_kernel(const float* __restrict__ init, float* __restrict__ state,
                                  unsigned short* __restrict__ Sb) {
  size_t i = (size_t)blockIdx.x * 256 + threadIdx.x;
  float v = init[i];
  state[i] = v;
  Sb[i] = f2bf(v);
}

__global__ void rowstats_kernel(const float* __restrict__ state, const float* __restrict__ target,
                                const float* __restrict__ linear, const float* __restrict__ weights,
                                float* __restrict__ lin, float* __restrict__ mixer,
                                float* __restrict__ pen, float* __restrict__ sum_old,
                                float* __restrict__ quad) {
  __shared__ float sm[256];
  int b = blockIdx.x, tid = threadIdx.x;
  const float* s = state + (size_t)b * DIM;
  const float* t = target + (size_t)b * DIM;
  float l = 0.f, m = 0.f, p = 0.f, so = 0.f;
  for (int d = tid; d < DIM; d += 256) {
    float v = s[d];
    l += v * linear[d];
    m += v * weights[d];
    float dd = v - t[d];
    p += dd * dd;
    so += v;
  }
  l  = block_reduce<256>(l, sm);
  m  = block_reduce<256>(m, sm);
  p  = block_reduce<256>(p, sm);
  so = block_reduce<256>(so, sm);
  if (tid == 0) {
    lin[b] = l; mixer[b] = m; pen[b] = p; sum_old[b] = so;
    quad[b] = 0.f;  // gemm_quad accumulates into this
  }
}

// Fused GEMM + row-dot: quad[b] += sum_n state[b,n] * (state@Q)[b,n]
// Block tile: M=32 x N=64; 8 waves, each one 16x16 WMMA accumulator; K chunks of 32.
__global__ void gemm_quad_kernel(const unsigned short* __restrict__ Sb,
                                 const unsigned short* __restrict__ Qt,
                                 const float* __restrict__ state,
                                 float* __restrict__ quad) {
  const int n0 = blockIdx.x * 64;
  const int row0 = blockIdx.y * 32;
  const int tid = threadIdx.x;
  const int wave = tid >> 5, lane = tid & 31;
  const int mi = wave >> 2, ni = wave & 3;   // 2 x 4 wave grid over 32x64 tile
  const int h = lane >> 4, l15 = lane & 15;
  (void)wave;

  __shared__ __align__(16) unsigned short Alds[32 * 32];  // [m][k]  2KB
  __shared__ __align__(16) unsigned short Blds[64 * 32];  // [n][k]  4KB (Qt rows)
  __shared__ float red[32];

  if (tid < 32) red[tid] = 0.f;
  __syncthreads();

  v8f acc = {};
  for (int k0 = 0; k0 < DIM; k0 += 32) {
#if HAS_TDM
    if (tid < 32) {
      unsigned a_off = (unsigned)(unsigned long long)(const void*)&Alds[0];
      unsigned b_off = (unsigned)(unsigned long long)(const void*)&Blds[0];
      tdm_load_2d(a_off, Sb + (size_t)row0 * DIM + k0, 32u, 32u, (unsigned)DIM, (unsigned)BATCH, (unsigned)DIM);
      tdm_load_2d(b_off, Qt + (size_t)n0 * DIM + k0, 32u, 64u, (unsigned)DIM, (unsigned)DIM, (unsigned)DIM);
      __builtin_amdgcn_s_wait_tensorcnt(0);
    }
    __syncthreads();
#else
    for (int i = tid; i < 32 * 32; i += 256)
      Alds[i] = Sb[(size_t)(row0 + (i >> 5)) * DIM + k0 + (i & 31)];
    for (int i = tid; i < 64 * 32; i += 256)
      Blds[i] = Qt[(size_t)(n0 + (i >> 5)) * DIM + k0 + (i & 31)];
    __syncthreads();
#endif
    // A fragment (16x32 bf16): lane row = mi*16+l15; v0..3: K=8h..8h+7, v4..7: K=16+8h..
    const unsigned short* arow = &Alds[(mi * 16 + l15) * 32];
    uint4 alo = *(const uint4*)(arow + 8 * h);
    uint4 ahi = *(const uint4*)(arow + 16 + 8 * h);
    // B fragment (32x16 bf16): lane col = ni*16+l15; v0..3: K=8h..8h+7, v4..7: K=16+8h..
    const unsigned short* brow = &Blds[(ni * 16 + l15) * 32];
    uint4 blo = *(const uint4*)(brow + 8 * h);
    uint4 bhi = *(const uint4*)(brow + 16 + 8 * h);

    v16bf a = mk_frag(alo, ahi);
    v16bf b = mk_frag(blo, bhi);
    acc = __builtin_amdgcn_wmma_f32_16x16x32_bf16(false, a, false, b, (short)0, acc, false, false);
    __syncthreads();  // protect LDS before next chunk's fill
  }

  // D layout: element (M = v + 8h, N = l15) within the 16x16 subtile.
  const int n_g = n0 + ni * 16 + l15;
#pragma unroll
  for (int v = 0; v < 8; ++v) {
    int m_loc = mi * 16 + v + 8 * h;
    float prod = acc[v] * state[(size_t)(row0 + m_loc) * DIM + n_g];
    unsafeAtomicAdd(&red[m_loc], prod);   // ds_add_f32
  }
  __syncthreads();
  if (tid < 32) unsafeAtomicAdd(&quad[row0 + tid], red[tid]);  // global_atomic_add_f32
}

__global__ void hamiltonian_kernel(const float* __restrict__ quad, const float* __restrict__ lin,
                                   const float* __restrict__ pen, const float* __restrict__ mixer,
                                   float ap, float* __restrict__ H, float* __restrict__ evo) {
  int b = blockIdx.x * 256 + threadIdx.x;
  if (b < BATCH) {
    float hh = (1.0f - ap) * mixer[b] + ap * (quad[b] + lin[b] + pen[b]);
    H[b] = hh;
    float x = hh * 0.01f;  // DT
    evo[b] = cosf(x) - sinf(x);
  }
}

__global__ void candidate_kernel(const float* __restrict__ state, const float* __restrict__ target,
                                 const float* __restrict__ evo, const float* __restrict__ namp,
                                 float one_m_ap, unsigned step,
                                 float* __restrict__ cand, float* __restrict__ sum_new,
                                 float* __restrict__ pen_cand) {
  __shared__ float sm[256];
  int b = blockIdx.x, tid = threadIdx.x;
  float e = evo[b];
  float ns = namp[0] * one_m_ap;
  float sn = 0.f, pc = 0.f;
  for (int d = tid; d < DIM; d += 256) {
    size_t idx = (size_t)b * DIM + d;
    float nz = gauss01(hash3(step, (unsigned)idx, 0x243F6A88u)) * ns;
    float c = e * state[idx] + nz;
    cand[idx] = c;
    sn += c;
    float dd = c - target[idx];
    pc += dd * dd;
  }
  sn = block_reduce<256>(sn, sm);
  pc = block_reduce<256>(pc, sm);
  if (tid == 0) { sum_new[b] = sn; pen_cand[b] = pc; }
}

__global__ void accept_kernel(const float* __restrict__ H, const float* __restrict__ sum_old,
                              const float* __restrict__ sum_new, const float* __restrict__ pen,
                              const float* __restrict__ pen_cand, float beta, unsigned step,
                              int* __restrict__ flags, float* __restrict__ energy_out) {
  __shared__ float sm[512];
  __shared__ int acc_s;
  int b = threadIdx.x;  // 512 threads, one per batch row
  float diff = H[b] * (sum_new[b] - sum_old[b]);
  float p = expf(-beta * fmaxf(diff, 0.f));
  float prob = block_reduce<512>(p, sm) * (1.0f / BATCH);
  if (b == 0) {
    float u = (float)hash3(step, 0xDEADBEEFu, 0x1234567u) * 2.3283064e-10f;
    int a = (u < prob) ? 1 : 0;
    acc_s = a;
    flags[0] = a;
  }
  __syncthreads();
  int a = acc_s;
  float e = sqrtf(a ? pen_cand[b] : pen[b]);
  float esum = block_reduce<512>(e, sm);
  if (b == 0) energy_out[step] = esum * (1.0f / BATCH);
}

__global__ void commit_kernel(float* __restrict__ state, const float* __restrict__ cand,
                              const int* __restrict__ flags, unsigned short* __restrict__ Sb) {
  size_t i = (size_t)blockIdx.x * 256 + threadIdx.x;
  float v = flags[0] ? cand[i] : state[i];
  state[i] = v;
  Sb[i] = f2bf(v);
}

__global__ void final_copy_kernel(const float* __restrict__ state, float* __restrict__ out) {
  size_t i = (size_t)blockIdx.x * 256 + threadIdx.x;
  out[i] = state[i];
}

// ---------------- host launcher ----------------

extern "C" void kernel_launch(void* const* d_in, const int* in_sizes, int n_in,
                              void* d_out, int out_size, void* d_ws, size_t ws_size,
                              hipStream_t stream) {
  const float* initial = (const float*)d_in[0];
  const float* target  = (const float*)d_in[1];
  const float* Q       = (const float*)d_in[2];
  const float* linear  = (const float*)d_in[3];
  const float* weights = (const float*)d_in[4];
  const float* namp    = (const float*)d_in[5];

  // Workspace layout (~54.6 MB)
  char* w = (char*)d_ws;
  const size_t QT_BYTES = (size_t)DIM * DIM * 2;     // 32 MB bf16 Q^T
  const size_t SB_BYTES = (size_t)BATCH * DIM * 2;   //  4 MB bf16 state
  const size_t ST_BYTES = (size_t)BATCH * DIM * 4;   //  8 MB fp32 state
  unsigned short* Qt   = (unsigned short*)w;
  unsigned short* Sb   = (unsigned short*)(w + QT_BYTES);
  float* state         = (float*)(w + QT_BYTES + SB_BYTES);
  float* cand          = (float*)(w + QT_BYTES + SB_BYTES + ST_BYTES);
  float* rows          = (float*)(w + QT_BYTES + SB_BYTES + 2 * ST_BYTES);
  float* quad     = rows + 0 * BATCH;
  float* lin      = rows + 1 * BATCH;
  float* mixer    = rows + 2 * BATCH;
  float* pen      = rows + 3 * BATCH;
  float* sum_old  = rows + 4 * BATCH;
  float* sum_new  = rows + 5 * BATCH;
  float* pen_cand = rows + 6 * BATCH;
  float* H        = rows + 7 * BATCH;
  float* evo      = rows + 8 * BATCH;
  int*   flags    = (int*)(rows + 9 * BATCH);

  float* out_state  = (float*)d_out;
  float* out_energy = out_state + (size_t)BATCH * DIM;

  const int EW_BLOCKS = (BATCH * DIM) / 256;  // 8192

  // One-time prep
  qt_convert_kernel<<<dim3(DIM / 32, DIM / 32), dim3(32, 8), 0, stream>>>(Q, Qt);
  init_state_kernel<<<EW_BLOCKS, 256, 0, stream>>>(initial, state, Sb);

  for (int step = 0; step < NSTEPS; ++step) {
    float sfrac = (float)step / (float)(NSTEPS - 1);
    float ap = 1.0f - expf(-5.0f * sfrac);
    float beta = 1.0f / (1.0f - ap + 1e-8f);

    rowstats_kernel<<<BATCH, 256, 0, stream>>>(state, target, linear, weights,
                                               lin, mixer, pen, sum_old, quad);
    gemm_quad_kernel<<<dim3(DIM / 64, BATCH / 32), 256, 0, stream>>>(Sb, Qt, state, quad);
    hamiltonian_kernel<<<BATCH / 256, 256, 0, stream>>>(quad, lin, pen, mixer, ap, H, evo);
    candidate_kernel<<<BATCH, 256, 0, stream>>>(state, target, evo, namp, 1.0f - ap,
                                                (unsigned)step, cand, sum_new, pen_cand);
    accept_kernel<<<1, 512, 0, stream>>>(H, sum_old, sum_new, pen, pen_cand, beta,
                                         (unsigned)step, flags, out_energy);
    commit_kernel<<<EW_BLOCKS, 256, 0, stream>>>(state, cand, flags, Sb);
  }

  final_copy_kernel<<<EW_BLOCKS, 256, 0, stream>>>(state, out_state);
}
